// BeamSearchLayer_14139032338498
// MI455X (gfx1250) — matrix operand order repaired and checked by reading, not compiled, not run on hardware
//
#include <hip/hip_runtime.h>
#include <hip/hip_bf16.h>
#include <math.h>

#define Bn 16
#define BEAMn 4
#define Vn 32000
#define Dn 512
#define Sn 128
#define Ln 33            // MAX_STEP + 1
#define MAX_STEPn 32
#define START_T 1
#define END_T 2
#define PAD_T 1
#define FMINF (-3.4028234663852886e38f)

typedef __attribute__((ext_vector_type(16))) __bf16 v16bf;
typedef __attribute__((ext_vector_type(8)))  float  v8f;
typedef __attribute__((ext_vector_type(4)))  int    v4i;

__device__ __forceinline__ unsigned short f2bf(float f) {
  union { float f; unsigned u; } x; x.f = f;
  return (unsigned short)(x.u >> 16);
}

// Swizzled A-fragment index for hidden (64 rows x 512 K, bf16).
// Fragment = (batch-group g, K-block kb): 512 halfwords laid out as
// [half0: lane*8 + j][half1: 256 + lane*8 + j] so each lane loads 2x b128.
// A 16x32 layout: u[j] -> k = kb*32 + hi*8 + j ; u[8+j] -> k = kb*32 + 16 + hi*8 + j ; lane = hi*16 + m.
__device__ __forceinline__ int a_sw_index(int row, int d) {
  int g = row >> 4, m = row & 15;
  int kb = d >> 5, o = d & 31;
  int half = o >> 4;        // 0: k-offset 0..15 (u[0..7]), 1: 16..31 (u[8..15])
  int oo = o & 15;
  int hi = oo >> 3;
  int j  = oo & 7;
  int L  = hi * 16 + m;
  return (g * 16 + kb) * 512 + half * 256 + L * 8 + j;
}

// ---- state buffer word offsets (4-byte words) ----
#define ST_SEQS  0
#define ST_LP    2112
#define ST_SC    2176
#define ST_FFL   2240
#define ST_FSC   2304
#define ST_FSEQ  2368
#define ST_WORDS 4480

// One-time: W_out (D x V f32, row-major) -> bf16 pre-swizzled into WMMA B fragments.
// Fragment = (v-tile vt, K-block kb): B 32x16 layout u[j] -> k = kb*32 + (j>>3)*16 + hi*8 + (j&7), n = vt*16 + (lane&15).
// Stored as [half0: lane*8+j][half1: 256 + lane*8+j] -> two dense b128 loads per fragment.
__global__ void k_convw(const float* __restrict__ W, unsigned short* __restrict__ Bsw, int n) {
  int idx = blockIdx.x * 256 + threadIdx.x;
  if (idx >= n) return;
  int frag = idx >> 9;          // /512
  int r    = idx & 511;
  int half = r >> 8;
  int L    = (r >> 3) & 31;
  int j    = half * 8 + (r & 7);
  int vt   = frag >> 4;
  int kb   = frag & 15;
  int col  = vt * 16 + (L & 15);
  int k    = kb * 32 + (j >> 3) * 16 + (L >> 4) * 8 + (j & 7);
  Bsw[idx] = f2bf(W[(size_t)k * Vn + col]);
}

// mem_ctx = tanh(mean(memory, S) @ W_mem); hidden(t=0) = emb[START] + mem_ctx (A-swizzled); init state.
__global__ void __launch_bounds__(256) k_init(const float* __restrict__ memory,
                                              const float* __restrict__ emb,
                                              const float* __restrict__ Wm,
                                              float* __restrict__ mem_ctx,
                                              unsigned short* __restrict__ hid,
                                              unsigned int* __restrict__ st0) {
  __shared__ float meanv[Dn];
  __shared__ float ctx[Dn];
  int b = blockIdx.x, tid = threadIdx.x;
  for (int d = tid; d < Dn; d += 256) {
    float s = 0.f;
    for (int j = 0; j < Sn; ++j) s += memory[(size_t)(b * Sn + j) * Dn + d];
    meanv[d] = s * (1.0f / Sn);
  }
  __syncthreads();
  for (int d = tid; d < Dn; d += 256) {
    float acc = 0.f;
    for (int j = 0; j < Dn; ++j) acc += meanv[j] * Wm[j * Dn + d];
    float v = tanhf(acc);
    ctx[d] = v;
    mem_ctx[b * Dn + d] = v;
  }
  __syncthreads();
  for (int i = tid; i < BEAMn * Dn; i += 256) {
    int k = i >> 9, d = i & (Dn - 1);
    hid[a_sw_index(b * BEAMn + k, d)] = f2bf(emb[(size_t)START_T * Dn + d] + ctx[d]);
  }
  if (b == 0) {
    int*   si = (int*)st0;
    float* sf = (float*)st0;
    for (int i = tid; i < Bn * BEAMn * Ln; i += 256) {
      si[ST_SEQS + i] = (i % Ln == 0) ? START_T : PAD_T;
      si[ST_FSEQ + i] = 0;
    }
    for (int i = tid; i < Bn * BEAMn; i += 256) {
      sf[ST_LP  + i] = (i % BEAMn == 0) ? 0.0f : FMINF;
      sf[ST_SC  + i] = 0.0f;
      si[ST_FFL + i] = 0;
      sf[ST_FSC + i] = FMINF;
    }
  }
}

// GEMM: logits[64 x 32000] = hidden(bf16) @ W(bf16), f32 accum via v_wmma_f32_16x16x32_bf16.
// Both operands pre-swizzled into fragment order: per K-block each wave issues
// 2 b128 loads for A (amortized over 2 N-tiles) + 2x2 b128 loads for B + 2 WMMAs.
// grid = (V/128, 4 batch-groups); 4 waves/block; wave owns two 16-col tiles.
__global__ void __launch_bounds__(128) k_gemm(const unsigned short* __restrict__ Bsw,
                                              const unsigned short* __restrict__ Asw,
                                              float* __restrict__ logits) {
  int g = blockIdx.y, tid = threadIdx.x;
  int lane = tid & 31, wv = tid >> 5;
  int vt0 = blockIdx.x * 8 + wv * 2;           // first of two 16-col tiles

  const v4i* Ap  = (const v4i*)(Asw + (size_t)g   * 16 * 512);
  const v4i* Bp0 = (const v4i*)(Bsw + (size_t)vt0 * 16 * 512);
  const v4i* Bp1 = (const v4i*)(Bsw + (size_t)(vt0 + 1) * 16 * 512);

  union U { v16bf v; v4i q[2]; };
  v8f c0 = {0.f, 0.f, 0.f, 0.f, 0.f, 0.f, 0.f, 0.f};
  v8f c1 = {0.f, 0.f, 0.f, 0.f, 0.f, 0.f, 0.f, 0.f};

  #pragma unroll
  for (int kb = 0; kb < Dn / 32; ++kb) {
    U a, b0, b1;
    int fo = kb * 64 + lane;                   // fragment = 64 x 16B
    a.q[0]  = Ap[fo];       a.q[1]  = Ap[fo + 32];
    b0.q[0] = Bp0[fo];      b0.q[1] = Bp0[fo + 32];
    b1.q[0] = Bp1[fo];      b1.q[1] = Bp1[fo + 32];
    c0 = __builtin_amdgcn_wmma_f32_16x16x32_bf16(false, a.v, false, b0.v,
                                                 (short)0, c0, false, false);
    c1 = __builtin_amdgcn_wmma_f32_16x16x32_bf16(false, a.v, false, b1.v,
                                                 (short)0, c1, false, false);
  }
  // C layout: VGPR i -> M=i (lanes 0-15) / M=i+8 (lanes 16-31), N = lane&15
  int nl = lane & 15, m0 = (lane >> 4) * 8;
  int col0 = vt0 * 16 + nl, col1 = col0 + 16;
  #pragma unroll
  for (int i = 0; i < 8; ++i) {
    logits[(size_t)(g * 16 + m0 + i) * Vn + col0] = c0[i];
    logits[(size_t)(g * 16 + m0 + i) * Vn + col1] = c1[i];
  }
}

// Per-step reduction + beam update. One block per batch; double-buffered state.
__global__ void __launch_bounds__(256) k_step(int t, const float* __restrict__ emb,
                                              const float* __restrict__ mem_ctx,
                                              unsigned short* __restrict__ hid,
                                              const float* __restrict__ logits,
                                              const unsigned int* __restrict__ stin,
                                              unsigned int* __restrict__ stout) {
  const int*   seqs_i = (const int*)stin + ST_SEQS;
  const float* lp_i   = (const float*)stin + ST_LP;
  const float* sc_i   = (const float*)stin + ST_SC;
  const int*   ffl_i  = (const int*)stin + ST_FFL;
  const float* fsc_i  = (const float*)stin + ST_FSC;
  const int*   fsq_i  = (const int*)stin + ST_FSEQ;
  int*   seqs_o = (int*)stout + ST_SEQS;
  float* lp_o   = (float*)stout + ST_LP;
  float* sc_o   = (float*)stout + ST_SC;
  int*   ffl_o  = (int*)stout + ST_FFL;
  float* fsc_o  = (float*)stout + ST_FSC;
  int*   fsq_o  = (int*)stout + ST_FSEQ;

  int b = blockIdx.x, tid = threadIdx.x;
  __shared__ float rv[256]; __shared__ int ri[256];
  __shared__ float s_lse[BEAMn], s_lp[BEAMn];
  __shared__ int s_active;
  __shared__ int s_pick[8]; __shared__ float s_pv[8];
  __shared__ int s_abeam[BEAMn], s_asym[BEAMn];
  __shared__ int s_fsel[BEAMn];
  __shared__ int s_cbeam[8], s_csym[8];

  // global 'active' scalar from pre-update state (race-free via double buffer)
  if (tid == 0) {
    bool all_gt = true;
    for (int bb = 0; bb < Bn; ++bb) {
      float ba = lp_i[bb * BEAMn + 0] / (float)MAX_STEPn;
      int any = 0; float wf = 0.f; bool first = true;
      for (int k = 0; k < BEAMn; ++k) {
        int fl = ffl_i[bb * BEAMn + k];
        any |= fl;
        float p = fsc_i[bb * BEAMn + k] * (float)fl;
        wf = first ? p : fminf(wf, p); first = false;
      }
      if (!any) wf += FMINF;
      all_gt = all_gt && (wf > ba);
    }
    s_active = all_gt ? 0 : 1;
  }
  __syncthreads();

  if (!s_active) {  // carry unchanged
    for (int i = tid; i < BEAMn * Ln; i += 256) {
      seqs_o[b * BEAMn * Ln + i] = seqs_i[b * BEAMn * Ln + i];
      fsq_o [b * BEAMn * Ln + i] = fsq_i [b * BEAMn * Ln + i];
    }
    if (tid < BEAMn) {
      lp_o [b * BEAMn + tid] = lp_i [b * BEAMn + tid];
      sc_o [b * BEAMn + tid] = sc_i [b * BEAMn + tid];
      ffl_o[b * BEAMn + tid] = ffl_i[b * BEAMn + tid];
      fsc_o[b * BEAMn + tid] = fsc_i[b * BEAMn + tid];
    }
    return;
  }

  // exact log-softmax stats per beam: max, then logsumexp
  for (int k = 0; k < BEAMn; ++k) {
    const float* row = logits + (size_t)(b * BEAMn + k) * Vn;
    float lm = -INFINITY;
    for (int v = tid; v < Vn; v += 256) lm = fmaxf(lm, row[v]);
    rv[tid] = lm; __syncthreads();
    for (int off = 128; off; off >>= 1) {
      if (tid < off) rv[tid] = fmaxf(rv[tid], rv[tid + off]);
      __syncthreads();
    }
    float m = rv[0]; __syncthreads();
    float ls = 0.f;
    for (int v = tid; v < Vn; v += 256) ls += expf(row[v] - m);
    rv[tid] = ls; __syncthreads();
    for (int off = 128; off; off >>= 1) {
      if (tid < off) rv[tid] += rv[tid + off];
      __syncthreads();
    }
    if (tid == 0) { s_lse[k] = m + logf(rv[0]); s_lp[k] = lp_i[b * BEAMn + k]; }
    __syncthreads();
  }

  // top-8 over 4*32000 scores, stable (tie -> lowest flat index) like lax.top_k
  float inv_pen = 1.0f / (float)(t + 1);
  for (int p = 0; p < 8; ++p) {
    float bv = -INFINITY; int bi = 0x7fffffff;
    for (int k = 0; k < BEAMn; ++k) {
      const float* row = logits + (size_t)(b * BEAMn + k) * Vn;
      float base = s_lp[k] - s_lse[k];
      for (int v = tid; v < Vn; v += 256) {
        int flat = k * Vn + v;
        bool skip = false;
        for (int q = 0; q < p; ++q) skip = skip || (s_pick[q] == flat);
        if (skip) continue;
        float sc = (base + row[v]) * inv_pen;
        if (sc > bv || (sc == bv && flat < bi)) { bv = sc; bi = flat; }
      }
    }
    rv[tid] = bv; ri[tid] = bi; __syncthreads();
    for (int off = 128; off; off >>= 1) {
      if (tid < off) {
        if (rv[tid + off] > rv[tid] ||
            (rv[tid + off] == rv[tid] && ri[tid + off] < ri[tid])) {
          rv[tid] = rv[tid + off]; ri[tid] = ri[tid + off];
        }
      }
      __syncthreads();
    }
    if (tid == 0) { s_pick[p] = ri[0]; s_pv[p] = rv[0]; }
    __syncthreads();
  }

  if (tid == 0) {
    float ts[8]; int beam[8], sym[8], fl[8];
    for (int j = 0; j < 8; ++j) {
      ts[j] = s_pv[j];
      beam[j] = s_pick[j] / Vn;
      sym[j]  = s_pick[j] % Vn;
      fl[j]   = (sym[j] == END_T) ? 1 : 0;
      s_cbeam[j] = beam[j]; s_csym[j] = sym[j];
    }
    // alive beams: top-4 of (score + flag*FMIN), stable
    float av[8]; int used[8];
    for (int j = 0; j < 8; ++j) { av[j] = ts[j] + (fl[j] ? FMINF : 0.0f); used[j] = 0; }
    float pen = (float)(t + 1);
    for (int i = 0; i < BEAMn; ++i) {
      int best = -1;
      for (int j = 0; j < 8; ++j)
        if (!used[j] && (best < 0 || av[j] > av[best])) best = j;
      used[best] = 1;
      s_abeam[i] = beam[best]; s_asym[i] = sym[best];
      sc_o[b * BEAMn + i] = av[best];
      lp_o[b * BEAMn + i] = av[best] * pen;
    }
    // finished pool: concat(old fin[4], step_fin[8]) -> top-4, stable
    float cv[12]; int cf[12], cu[12];
    for (int j = 0; j < 4; ++j) { cv[j] = fsc_i[b * BEAMn + j]; cf[j] = ffl_i[b * BEAMn + j]; cu[j] = 0; }
    for (int j = 0; j < 8; ++j) { cv[4 + j] = ts[j] + (fl[j] ? 0.0f : FMINF); cf[4 + j] = fl[j]; cu[4 + j] = 0; }
    for (int i = 0; i < BEAMn; ++i) {
      int best = -1;
      for (int j = 0; j < 12; ++j)
        if (!cu[j] && (best < 0 || cv[j] > cv[best])) best = j;
      cu[best] = 1;
      s_fsel[i] = best;
      fsc_o[b * BEAMn + i] = cv[best];
      ffl_o[b * BEAMn + i] = cf[best];
    }
  }
  __syncthreads();

  // parallel sequence gather/scatter
  for (int idx = tid; idx < BEAMn * Ln; idx += 256) {
    int i = idx / Ln, j = idx % Ln;
    int av = (j == t + 1) ? s_asym[i] : seqs_i[(b * BEAMn + s_abeam[i]) * Ln + j];
    seqs_o[(b * BEAMn + i) * Ln + j] = av;
    int src = s_fsel[i], fv;
    if (src < 4) fv = (j == t + 1) ? PAD_T : fsq_i[(b * BEAMn + src) * Ln + j];
    else {
      int jj = src - 4;
      fv = (j == t + 1) ? s_csym[jj] : seqs_i[(b * BEAMn + s_cbeam[jj]) * Ln + j];
    }
    fsq_o[(b * BEAMn + i) * Ln + j] = fv;
  }
  // next-step hidden = emb[new last token] + mem_ctx, bf16, written in A-fragment order
  for (int idx = tid; idx < BEAMn * Dn; idx += 256) {
    int k = idx >> 9, d = idx & (Dn - 1);
    int tok = s_asym[k];
    hid[a_sw_index(b * BEAMn + k, d)] = f2bf(emb[(size_t)tok * Dn + d] + mem_ctx[b * Dn + d]);
  }
}

__global__ void k_fin(const unsigned int* __restrict__ st, float* __restrict__ out) {
  const int*   seqs = (const int*)st + ST_SEQS;
  const float* sc   = (const float*)st + ST_SC;
  const int*   ffl  = (const int*)st + ST_FFL;
  const float* fsc  = (const float*)st + ST_FSC;
  const int*   fsq  = (const int*)st + ST_FSEQ;
  int tid = threadIdx.x;
  for (int idx = tid; idx < Bn * BEAMn * Ln; idx += 256) {
    int b = idx / (BEAMn * Ln);
    int any = ffl[b * BEAMn] | ffl[b * BEAMn + 1] | ffl[b * BEAMn + 2] | ffl[b * BEAMn + 3];
    out[idx] = (float)(any ? fsq[idx] : seqs[idx]);
  }
  for (int idx = tid; idx < Bn * BEAMn; idx += 256) {
    int b = idx / BEAMn;
    int any = ffl[b * BEAMn] | ffl[b * BEAMn + 1] | ffl[b * BEAMn + 2] | ffl[b * BEAMn + 3];
    out[Bn * BEAMn * Ln + idx] = any ? fsc[idx] : sc[idx];
  }
}

extern "C" void kernel_launch(void* const* d_in, const int* in_sizes, int n_in,
                              void* d_out, int out_size, void* d_ws, size_t ws_size,
                              hipStream_t stream) {
  (void)in_sizes; (void)n_in; (void)out_size; (void)ws_size;
  const float* memory = (const float*)d_in[0];
  const float* emb    = (const float*)d_in[1];
  const float* W_mem  = (const float*)d_in[2];
  const float* W_out  = (const float*)d_in[3];

  char* ws = (char*)d_ws;
  size_t off = 0;
  unsigned short* Bsw = (unsigned short*)(ws + off); off += (size_t)Dn * Vn * 2;        // 32.8 MB
  float* mem_ctx      = (float*)(ws + off);          off += (size_t)Bn * Dn * 4;        // 32 KB
  unsigned short* hid = (unsigned short*)(ws + off); off += (size_t)Bn * BEAMn * Dn * 2;// 64 KB (A-swizzled)
  float* logits       = (float*)(ws + off);          off += (size_t)Bn * BEAMn * Vn * 4;// 8.2 MB
  unsigned int* st0   = (unsigned int*)(ws + off);   off += (size_t)ST_WORDS * 4;
  unsigned int* st1   = (unsigned int*)(ws + off);   off += (size_t)ST_WORDS * 4;

  k_convw<<<(Dn * Vn + 255) / 256, 256, 0, stream>>>(W_out, Bsw, Dn * Vn);
  k_init <<<Bn, 256, 0, stream>>>(memory, emb, W_mem, mem_ctx, hid, st0);

  unsigned int* sbuf[2] = { st0, st1 };
  for (int t = 0; t < MAX_STEPn; ++t) {
    k_gemm<<<dim3(Vn / 128, Bn / 4), 128, 0, stream>>>(Bsw, hid, logits);
    k_step<<<Bn, 256, 0, stream>>>(t, emb, mem_ctx, hid, logits, sbuf[t & 1], sbuf[(t + 1) & 1]);
  }
  // after 32 steps, final state lands back in st0
  k_fin<<<1, 256, 0, stream>>>(st0, (float*)d_out);
}
